// Encoder_12979391169239
// MI455X (gfx1250) — compile-verified, matrix operand order checked by reading, not run on hardware
//
#include <hip/hip_runtime.h>

#define B_ 4
#define S_ 512
#define D_ 128
#define H_ 128

typedef __attribute__((ext_vector_type(16))) __bf16 v16bf;
typedef __attribute__((ext_vector_type(8)))  float  v8f;

// K-dimension offset for packed bf16 pair p (0..7) in a 16x32 A / 32x16 B
// fragment, lane group g = lane/16 (ISA 7.12.2 16-bit layouts):
// VGPR0..3 hold K={0..7}+8g, VGPR4..7 hold K={16..23}+8g.
__device__ __forceinline__ int kpair_off(int p, int g) {
  return ((p < 4) ? (2 * p) : (16 + 2 * (p - 4))) + 8 * g;
}

// ---------------------------------------------------------------------------
// Kernel A: Q/K/V projections.  out[m, n] = sum_d x[m,d] * W[n,d] + bias[n]
// One wave per 16x16 output tile, 4 chained v_wmma_f32_16x16x32_bf16 over D.
// ---------------------------------------------------------------------------
__global__ __launch_bounds__(32)
void qkv_proj(const float* __restrict__ x,
              const float* __restrict__ Qw, const float* __restrict__ Qbias,
              const float* __restrict__ Kw, const float* __restrict__ Kbias,
              const float* __restrict__ Vw, const float* __restrict__ Vbias,
              float* __restrict__ Qo, float* __restrict__ Ko, float* __restrict__ Vo)
{
  const int lane = threadIdx.x;
  const int g  = lane >> 4;
  const int ln = lane & 15;
  const int m0 = blockIdx.x * 16;   // row tile in [B*S]
  const int n0 = blockIdx.y * 16;   // col tile in [H]
  const int z  = blockIdx.z;        // 0=Q 1=K 2=V

  const float* W    = (z == 0) ? Qw    : (z == 1) ? Kw    : Vw;
  const float* bias = (z == 0) ? Qbias : (z == 1) ? Kbias : Vbias;
  float*       out  = (z == 0) ? Qo    : (z == 1) ? Ko    : Vo;

  const float* arow = x + (size_t)(m0 + ln) * D_;  // A row  (x row-major)
  const float* brow = W + (size_t)(n0 + ln) * D_;  // B col  (= W row, since y = x W^T)

  v8f acc = {};
  for (int k0 = 0; k0 < D_; k0 += 32) {
    v16bf a, b;
#pragma unroll
    for (int p = 0; p < 8; ++p) {
      const int k = k0 + kpair_off(p, g);
      const float2 av = *(const float2*)(arow + k);
      const float2 bv = *(const float2*)(brow + k);
      a[2 * p] = (__bf16)av.x;  a[2 * p + 1] = (__bf16)av.y;
      b[2 * p] = (__bf16)bv.x;  b[2 * p + 1] = (__bf16)bv.y;
    }
    acc = __builtin_amdgcn_wmma_f32_16x16x32_bf16(false, a, false, b,
                                                  (short)0, acc, false, false);
  }
#pragma unroll
  for (int v = 0; v < 8; ++v) {
    const int m = m0 + v + 8 * g;   // C layout: VGPR v -> row v (+8 for lanes>=16)
    const int n = n0 + ln;
    out[(size_t)m * H_ + n] = acc[v] + bias[n];
  }
}

// ---------------------------------------------------------------------------
// Kernel B: fused attention for one (batch b, 16-row q tile).
//   score[q,k] = sum_h Q[q,h] * (K[k,h] + T[b,q,k,h])     (fused QK^T + bias)
//   softmax over k with masking, then out = P @ V.
// 8 waves: each streams a 64-wide k chunk of T exactly once.
// ---------------------------------------------------------------------------
__global__ __launch_bounds__(256)
void attn_kernel(const float* __restrict__ T,
                 const unsigned char* __restrict__ time_mask,
                 const unsigned char* __restrict__ attn_mask,
                 const float* __restrict__ Qb, const float* __restrict__ Kb,
                 const float* __restrict__ Vb,
                 float* __restrict__ out)
{
  __shared__ float sQ[16 * H_];      // Q tile, fp32
  __shared__ float sS[16][S_];       // raw scores -> probabilities (in place)
  __shared__ float sRed[16][16];
  __shared__ float sMax[16];
  __shared__ float sSum[16];

  const int tid  = threadIdx.x;
  const int lane = tid & 31;
  const int wave = tid >> 5;         // 0..7
  const int g    = lane >> 4;
  const int ln   = lane & 15;
  const int q0   = blockIdx.x * 16;
  const int b    = blockIdx.y;

  // ---- stage the Q tile (16 x 128 fp32) into LDS ----
  for (int i = tid; i < 16 * H_; i += 256)
    sQ[i] = Qb[((size_t)(b * S_) + q0 + (i >> 7)) * H_ + (i & 127)];
  __syncthreads();

  // ---- phase 1: fused scores; this wave owns k in [wave*64, wave*64+64) ----
  const int kw0 = wave * 64;
  const float* Kbase = Kb + (size_t)(b * S_) * H_;
  for (int q = 0; q < 16; ++q) {
    // B fragments: Q[b, q0+q, :] replicated across all 16 columns
    v16bf qf[4];
#pragma unroll
    for (int hs = 0; hs < 4; ++hs)
#pragma unroll
      for (int p = 0; p < 8; ++p) {
        const int h = hs * 32 + kpair_off(p, g);
        qf[hs][2 * p]     = (__bf16)sQ[q * H_ + h];
        qf[hs][2 * p + 1] = (__bf16)sQ[q * H_ + h + 1];
      }
    const float* Tq = T + ((size_t)(b * S_ + q0 + q)) * S_ * H_;
    for (int kt = 0; kt < 4; ++kt) {
      const int kbase = kw0 + kt * 16;
      const float* Trow = Tq    + (size_t)(kbase + ln) * H_;  // A row = k index
      const float* Krow = Kbase + (size_t)(kbase + ln) * H_;
      v8f acc = {};
#pragma unroll
      for (int hs = 0; hs < 4; ++hs) {
        v16bf a;
#pragma unroll
        for (int p = 0; p < 8; ++p) {
          const int h = hs * 32 + kpair_off(p, g);
          const float2 tv = *(const float2*)(Trow + h);
          const float2 kv = *(const float2*)(Krow + h);
          a[2 * p]     = (__bf16)(tv.x + kv.x);
          a[2 * p + 1] = (__bf16)(tv.y + kv.y);
        }
        acc = __builtin_amdgcn_wmma_f32_16x16x32_bf16(false, a, false, qf[hs],
                                                      (short)0, acc, false, false);
      }
      // every column of D is identical; lanes 0/16 hold k rows 0..7 / 8..15
      if (ln == 0) {
#pragma unroll
        for (int v = 0; v < 8; ++v)
          sS[q][kbase + v + 8 * g] = acc[v];
      }
    }
  }
  __syncthreads();

  // ---- phase 2: scale + mask + exact softmax (16 threads per row) ----
  const int qr = tid >> 4;
  const int lr = tid & 15;
  const unsigned char* tmb = time_mask + b * S_;
  const bool tmq = tmb[q0 + qr] != 0;
  const float rs = 0.08838834764831845f;  // 1/sqrt(128)
  float lmax = -3.4e38f;
  for (int i = 0; i < 32; ++i) {
    const int k = lr + 16 * i;
    const bool m = tmq || (tmb[k] != 0) || (attn_mask[(size_t)(q0 + qr) * S_ + k] != 0);
    const float s = m ? -4294967295.0f : sS[qr][k] * rs;
    sS[qr][k] = s;
    lmax = fmaxf(lmax, s);
  }
  sRed[qr][lr] = lmax;
  __syncthreads();
  if (lr == 0) {
    float m = sRed[qr][0];
#pragma unroll
    for (int j = 1; j < 16; ++j) m = fmaxf(m, sRed[qr][j]);
    sMax[qr] = m;
  }
  __syncthreads();
  const float rmax = sMax[qr];
  float lsum = 0.f;
  for (int i = 0; i < 32; ++i) {
    const int k = lr + 16 * i;
    const float p = __expf(sS[qr][k] - rmax);
    sS[qr][k] = p;
    lsum += p;
  }
  sRed[qr][lr] = lsum;
  __syncthreads();
  if (lr == 0) {
    float s = 0.f;
#pragma unroll
    for (int j = 0; j < 16; ++j) s += sRed[qr][j];
    sSum[qr] = s;
  }
  __syncthreads();

  // ---- phase 3: out[16, H] = P[16, 512] @ V[512, H]; wave owns 16 columns ----
  const int h0 = wave * 16;
  const float* Vbase = Vb + (size_t)(b * S_) * H_;
  v8f acc = {};
  for (int ks = 0; ks < S_; ks += 32) {
    v16bf a, bf;
#pragma unroll
    for (int p = 0; p < 8; ++p) {
      const int kk = ks + kpair_off(p, g);
      a[2 * p]     = (__bf16)sS[ln][kk];
      a[2 * p + 1] = (__bf16)sS[ln][kk + 1];
      bf[2 * p]     = (__bf16)Vbase[(size_t)kk * H_ + h0 + ln];
      bf[2 * p + 1] = (__bf16)Vbase[(size_t)(kk + 1) * H_ + h0 + ln];
    }
    acc = __builtin_amdgcn_wmma_f32_16x16x32_bf16(false, a, false, bf,
                                                  (short)0, acc, false, false);
  }
#pragma unroll
  for (int v = 0; v < 8; ++v) {
    const int q = v + 8 * g;
    out[((size_t)(b * S_) + q0 + q) * H_ + h0 + ln] = acc[v] / sSum[q];
  }
}

// ---------------------------------------------------------------------------
extern "C" void kernel_launch(void* const* d_in, const int* in_sizes, int n_in,
                              void* d_out, int out_size, void* d_ws, size_t ws_size,
                              hipStream_t stream)
{
  (void)in_sizes; (void)n_in; (void)out_size; (void)ws_size;
  const float*         x         = (const float*)d_in[0];
  const unsigned char* time_mask = (const unsigned char*)d_in[1];
  const unsigned char* attn_mask = (const unsigned char*)d_in[2];
  const float*         T         = (const float*)d_in[3];
  const float* Qw = (const float*)d_in[4];
  const float* Qc = (const float*)d_in[5];
  const float* Kw = (const float*)d_in[6];
  const float* Kc = (const float*)d_in[7];
  const float* Vw = (const float*)d_in[8];
  const float* Vc = (const float*)d_in[9];

  float* Qo = (float*)d_ws;                       // [B*S, H] fp32
  float* Ko = Qo + (size_t)B_ * S_ * H_;
  float* Vo = Ko + (size_t)B_ * S_ * H_;

  dim3 gA(B_ * S_ / 16, H_ / 16, 3);
  qkv_proj<<<gA, 32, 0, stream>>>(x, Qw, Qc, Kw, Kc, Vw, Vc, Qo, Ko, Vo);

  dim3 gB(S_ / 16, B_);
  attn_kernel<<<gB, 256, 0, stream>>>(T, time_mask, attn_mask, Qo, Ko, Vo,
                                      (float*)d_out);
}
